// LPVModel4_6820408066390
// MI455X (gfx1250) — compile-verified, matrix operand order checked by reading, not run on hardware
//
#include <hip/hip_runtime.h>
#include <hip/hip_bf16.h>
#include <math.h>

// ---------------------------------------------------------------------------
// LPV model pipeline for MI455X (gfx1250), wave32 + WMMA.
//
// Stages:
//   1. ctxt_kernel     : mean-pool + tanh(mean @ W_ctxt + b)          (VALU, tiny)
//   2. gemm_wmma       : A_ct = h @ W_A + b_A        [256 x 65536]    (bf16-split WMMA)
//   3. gemm_wmma       : B_mat = h @ W_B + b_B       [256 x 8192]     (bf16-split WMMA)
//   4. gemm_wmma       : P = F @ F  (F = 0.005*A_ct, batched 256x)    (bf16-split WMMA)
//   5. gemm_wmma mode1 : A_dt = I + 2F + 2P + 2*(P @ F)  (Neumann series for
//                        the Cayley solve; ||F|| ~ 2e-3 so truncation error
//                        ~2e-11 << f32 eps)                           (bf16-split WMMA)
//   6. rollout_kernel  : z_{k+1} = A_dt z_k + B u_k, A_dt resident in LDS
//                        (290 KB of the 320 KB WGP LDS)               (VALU matvec)
//   7. gemm_wmma       : x_traj   = z_traj @ W_dec + b_dec
//   8. gt_copy_kernel  : x_traj_gt = concat(x_init, x_future)
//   9. gemm_wmma       : z_traj_re = x_traj    @ W_enc + b_enc
//  10. gemm_wmma       : z_traj_gt = x_traj_gt @ W_enc + b_enc
//
// Precision: every WMMA GEMM uses a 2-term bf16 split of both operands and
// accumulates hi*hi + hi*lo + lo*hi in f32 (tf32x3-style, ~f32 accuracy) at
// ~3x the throughput of V_WMMA_F32_16X16X4_F32.
//
// Workspace: h (512 KB) + P (64 MB) + A_dt (64 MB) ~= 129 MB.
// ---------------------------------------------------------------------------

typedef __attribute__((ext_vector_type(16))) __bf16 v16bf;
typedef __attribute__((ext_vector_type(8)))  float  v8f;

struct U8 { unsigned int u[8]; };

#define BM   128
#define BN   128
#define BKC  32
#define LDSP 34   // padded row stride (shorts): 17 dwords -> gcd(17,64)=1, conflict-free

// K index of the low element of 32-bit pair v for this half-wave (ISA 7.12.2,
// 16-bit A-matrix 16x32 layout; B assumed symmetric as B^T).
__device__ __forceinline__ int kmap(int v, int hf) {
  return ((v & 4) << 2) + (hf << 3) + ((v & 3) << 1);
}

// Split f32 into bf16 hi (truncate) + bf16 lo of the residual.
__device__ __forceinline__ void split_store(unsigned short* hi, unsigned short* lo,
                                            int idx, float v) {
  unsigned int u = __float_as_uint(v);
  unsigned short h = (unsigned short)(u >> 16);
  float fh = __uint_as_float(((unsigned int)h) << 16);
  float fl = v - fh;
  hi[idx] = h;
  lo[idx] = (unsigned short)(__float_as_uint(fl) >> 16);
}

// Generic batched GEMM: C = op(A*scaleA @ B*scaleB), row-major.
// mode 0: C = acc + bias[col]          (bias may be null)
// mode 1: C = I + 2*fscale*F + 2*P + 2*acc   (Cayley/Neumann assembly)
__global__ __launch_bounds__(256) void gemm_wmma(
    const float* __restrict__ Ap, long long sA, int lda,
    const float* __restrict__ Bp, long long sB, int ldb,
    float* __restrict__ Cp, long long sC, int ldc,
    const float* __restrict__ bias,
    int M, int N, int K,
    float scaleA, float scaleB,
    int mode,
    const float* __restrict__ Fp, long long sF, float fscale,
    const float* __restrict__ Pp, long long sP)
{
  __shared__ unsigned short As_hi[BM * LDSP];
  __shared__ unsigned short As_lo[BM * LDSP];
  __shared__ unsigned short Bs_hi[BN * LDSP];
  __shared__ unsigned short Bs_lo[BN * LDSP];

  const int tid  = threadIdx.x;
  const int wave = tid >> 5;
  const int lane = tid & 31;
  const int hf   = lane >> 4;
  const int l16  = lane & 15;
  const int bz   = blockIdx.z;

  const float* Ab = Ap + (long long)bz * sA;
  const float* Bb = Bp + (long long)bz * sB;
  float*       Cb = Cp + (long long)bz * sC;

  const int m0 = blockIdx.y * BM;
  const int n0 = blockIdx.x * BN;

  v8f zero = {};
  v8f acc[8];
#pragma unroll
  for (int i = 0; i < 8; ++i) acc[i] = zero;

  const int nchunks = (K + BKC - 1) / BKC;
  for (int ck = 0; ck < nchunks; ++ck) {
    const int k0 = ck * BKC;
    {
      // A tile: 128x32, 16 elems/thread, coalesced 32-wide rows.
      const int c  = tid & 31;
      const int rb = tid >> 5;
#pragma unroll
      for (int i = 0; i < 16; ++i) {
        const int r  = rb + (i << 3);
        const int gr = m0 + r, gk = k0 + c;
        float v = 0.f;
        if (gr < M && gk < K) v = Ab[(long long)gr * lda + gk] * scaleA;
        split_store(As_hi, As_lo, r * LDSP + c, v);
      }
      // B tile: 32x128 stored transposed as Bs[n][kk], coalesced 128-wide rows.
      const int n  = tid & 127;
      const int kb = tid >> 7;
#pragma unroll
      for (int i = 0; i < 16; ++i) {
        const int kk = kb + (i << 1);
        const int gk = k0 + kk, gc = n0 + n;
        float v = 0.f;
        if (gk < K && gc < N) v = Bb[(long long)gk * ldb + gc] * scaleB;
        split_store(Bs_hi, Bs_lo, n * LDSP + kk, v);
      }
    }
    __syncthreads();

    // A fragments for this wave's 16-row strip.
    U8 ta_hi, ta_lo;
#pragma unroll
    for (int v = 0; v < 8; ++v) {
      const int kk = kmap(v, hf);
      const int r  = (wave << 4) + l16;
      ta_hi.u[v] = *(const unsigned int*)&As_hi[r * LDSP + kk];
      ta_lo.u[v] = *(const unsigned int*)&As_lo[r * LDSP + kk];
    }
    const v16bf a_hi = __builtin_bit_cast(v16bf, ta_hi);
    const v16bf a_lo = __builtin_bit_cast(v16bf, ta_lo);

#pragma unroll
    for (int nt = 0; nt < 8; ++nt) {
      U8 tb_hi, tb_lo;
#pragma unroll
      for (int v = 0; v < 8; ++v) {
        const int kk = kmap(v, hf);
        const int n  = (nt << 4) + l16;
        tb_hi.u[v] = *(const unsigned int*)&Bs_hi[n * LDSP + kk];
        tb_lo.u[v] = *(const unsigned int*)&Bs_lo[n * LDSP + kk];
      }
      const v16bf b_hi = __builtin_bit_cast(v16bf, tb_hi);
      const v16bf b_lo = __builtin_bit_cast(v16bf, tb_lo);
      // hi*hi + hi*lo + lo*hi : ~f32-accurate product, f32 accumulate.
      acc[nt] = __builtin_amdgcn_wmma_f32_16x16x32_bf16(false, a_hi, false, b_hi,
                                                        (short)0, acc[nt], false, false);
      acc[nt] = __builtin_amdgcn_wmma_f32_16x16x32_bf16(false, a_hi, false, b_lo,
                                                        (short)0, acc[nt], false, false);
      acc[nt] = __builtin_amdgcn_wmma_f32_16x16x32_bf16(false, a_lo, false, b_hi,
                                                        (short)0, acc[nt], false, false);
    }
    __syncthreads();
  }

  // Epilogue (C/D layout: row = v + 8*half, col = lane&15 per ISA 7.12.2).
#pragma unroll
  for (int nt = 0; nt < 8; ++nt) {
#pragma unroll
    for (int v = 0; v < 8; ++v) {
      const int gr = m0 + (wave << 4) + v + (hf << 3);
      const int gc = n0 + (nt << 4) + l16;
      if (gr < M && gc < N) {
        float r = acc[nt][v];
        const long long idx = (long long)gr * ldc + gc;
        if (mode == 0) {
          if (bias) r += bias[gc];
          Cb[idx] = r;
        } else {
          const float f = fscale * (Fp + (long long)bz * sF)[idx];
          const float p = (Pp + (long long)bz * sP)[idx];
          Cb[idx] = (gr == gc ? 1.f : 0.f) + 2.f * f + 2.f * p + 2.f * r;
        }
      }
    }
  }
}

// h = tanh(mean_t(concat(x_hist, u_hist)) @ W_ctxt + b_ctxt)
__global__ __launch_bounds__(128) void ctxt_kernel(
    const float* __restrict__ xh, const float* __restrict__ uh,
    const float* __restrict__ W, const float* __restrict__ bias,
    float* __restrict__ h)
{
  __shared__ float mean[96];
  const int b = blockIdx.x, t = threadIdx.x;
  if (t < 96) {
    float s = 0.f;
    if (t < 64) {
      for (int k = 0; k < 32; ++k) s += xh[((long long)b * 32 + k) * 64 + t];
    } else {
      for (int k = 0; k < 32; ++k) s += uh[((long long)b * 32 + k) * 32 + (t - 64)];
    }
    mean[t] = s * (1.f / 32.f);
  }
  __syncthreads();
  for (int j = t; j < 512; j += 128) {
    float a = bias[j];
#pragma unroll 8
    for (int i = 0; i < 96; ++i) a += mean[i] * W[i * 512 + j];
    h[(long long)b * 512 + j] = tanhf(a);
  }
}

// x_traj_gt = concat(x_init[:,None,:], x_future)
__global__ void gt_copy_kernel(const float* __restrict__ xi,
                               const float* __restrict__ xf,
                               float* __restrict__ out)
{
  const int idx = blockIdx.x * blockDim.x + threadIdx.x;
  if (idx >= 256 * 129 * 64) return;
  const int c = idx & 63;
  const int k = (idx >> 6) % 129;
  const int b = idx / (129 * 64);
  out[idx] = (k == 0) ? xi[b * 64 + c]
                      : xf[((long long)b * 128 + (k - 1)) * 64 + c];
}

// One workgroup per batch. A_dt[b] (256x256 f32, column-major + pad) lives in
// LDS for all 128 sequential steps; z stays in LDS; writes z_traj[b,k,:].
__global__ __launch_bounds__(256) void rollout_kernel(
    const float* __restrict__ Adt, const float* __restrict__ Bmat,
    const float* __restrict__ x_init, const float* __restrict__ u_future,
    const float* __restrict__ W_enc, const float* __restrict__ b_enc,
    float* __restrict__ z_traj)
{
  extern __shared__ float lds[];
  float* Acm = lds;                 // [j*257 + i] = A[i][j], pad -> conflict-free
  float* Bcm = Acm + 256 * 257;     // [c*257 + i] = B[i][c]
  float* zs  = Bcm + 32 * 257;      // [256]
  float* us  = zs + 256;            // [32]

  const int b = blockIdx.x;
  const int t = threadIdx.x;

  // Load A transposed: coalesced global reads, conflict-free LDS writes.
  for (int it = 0; it < 256; ++it)
    Acm[t * 257 + it] = Adt[(long long)b * 65536 + (long long)it * 256 + t];
  for (int it = 0; it < 32; ++it) {
    const int idx = it * 256 + t;
    const int i = idx >> 5, c = idx & 31;
    Bcm[c * 257 + i] = Bmat[(long long)b * 8192 + idx];
  }

  // z0 = enc(x_init)
  float z0 = b_enc[t];
#pragma unroll 8
  for (int c = 0; c < 64; ++c) z0 += x_init[b * 64 + c] * W_enc[c * 256 + t];
  zs[t] = z0;
  z_traj[(long long)b * (129 * 256) + t] = z0;
  __syncthreads();

  for (int k = 0; k < 128; ++k) {
    if (t < 32) us[t] = u_future[(long long)b * (128 * 32) + k * 32 + t];
    __syncthreads();
    float a = 0.f;
#pragma unroll 8
    for (int j = 0; j < 256; ++j) a += Acm[j * 257 + t] * zs[j];
#pragma unroll
    for (int c = 0; c < 32; ++c) a += Bcm[c * 257 + t] * us[c];
    __syncthreads();
    zs[t] = a;
    z_traj[(long long)b * (129 * 256) + (long long)(k + 1) * 256 + t] = a;
  }
}

extern "C" void kernel_launch(void* const* d_in, const int* in_sizes, int n_in,
                              void* d_out, int out_size, void* d_ws, size_t ws_size,
                              hipStream_t stream)
{
  (void)in_sizes; (void)n_in; (void)out_size; (void)ws_size;

  const float* x_history = (const float*)d_in[0];
  const float* u_history = (const float*)d_in[1];
  const float* x_init    = (const float*)d_in[2];
  const float* x_future  = (const float*)d_in[3];
  const float* u_future  = (const float*)d_in[4];
  // d_in[5] = n_steps (K = 128, fixed by the shapes)
  const float* W_ctxt = (const float*)d_in[6];
  const float* b_ctxt = (const float*)d_in[7];
  const float* W_A    = (const float*)d_in[8];
  const float* b_A    = (const float*)d_in[9];
  const float* W_B    = (const float*)d_in[10];
  const float* b_B    = (const float*)d_in[11];
  const float* W_enc  = (const float*)d_in[12];
  const float* b_enc  = (const float*)d_in[13];
  const float* W_dec  = (const float*)d_in[14];
  const float* b_dec  = (const float*)d_in[15];

  // Output tuple, flat in return order.
  float* out    = (float*)d_out;
  float* z_traj = out;              // [256,129,256]
  float* x_traj = out + 8454144;    // [256,129, 64]
  float* A_ct   = out + 10567680;   // [256,256,256]
  float* B_mat  = out + 27344896;   // [256,256, 32]
  float* z_re   = out + 29442048;   // [256,129,256]
  float* x_gt   = out + 37896192;   // [256,129, 64]
  float* z_gt   = out + 40009728;   // [256,129,256]

  // Workspace: h | P=F^2 | A_dt  (~129 MB)
  float* ws    = (float*)d_ws;
  float* h_buf = ws;                    // 256*512
  float* P_buf = ws + 131072;           // 256*256*256
  float* Adt   = P_buf + 16777216;      // 256*256*256

  const float FS = 0.5f * 0.01f;        // dt/2

  // 1. context encoder
  ctxt_kernel<<<256, 128, 0, stream>>>(x_history, u_history, W_ctxt, b_ctxt, h_buf);

  // 2. A_ct = h @ W_A + b_A   (M=256, K=512, N=65536)
  gemm_wmma<<<dim3(512, 2, 1), 256, 0, stream>>>(
      h_buf, 0, 512,  W_A, 0, 65536,  A_ct, 0, 65536,  b_A,
      256, 65536, 512, 1.f, 1.f, 0, nullptr, 0, 0.f, nullptr, 0);

  // 3. B_mat = h @ W_B + b_B  (M=256, K=512, N=8192)
  gemm_wmma<<<dim3(64, 2, 1), 256, 0, stream>>>(
      h_buf, 0, 512,  W_B, 0, 8192,  B_mat, 0, 8192,  b_B,
      256, 8192, 512, 1.f, 1.f, 0, nullptr, 0, 0.f, nullptr, 0);

  // 4. P = F @ F, batched (F = FS * A_ct)
  gemm_wmma<<<dim3(2, 2, 256), 256, 0, stream>>>(
      A_ct, 65536, 256,  A_ct, 65536, 256,  P_buf, 65536, 256,  nullptr,
      256, 256, 256, FS, FS, 0, nullptr, 0, 0.f, nullptr, 0);

  // 5. A_dt = I + 2F + 2P + 2*(P @ F), batched  (Neumann/Cayley)
  gemm_wmma<<<dim3(2, 2, 256), 256, 0, stream>>>(
      P_buf, 65536, 256,  A_ct, 65536, 256,  Adt, 65536, 256,  nullptr,
      256, 256, 256, 1.f, FS, 1, A_ct, 65536, FS, P_buf, 65536);

  // 6. rollout: z_{k+1} = A_dt z_k + B u_k, A_dt resident in LDS (~290 KB/WGP)
  const size_t roll_lds = (size_t)(256 * 257 + 32 * 257 + 256 + 32) * 4;
  rollout_kernel<<<256, 256, roll_lds, stream>>>(
      Adt, B_mat, x_init, u_future, W_enc, b_enc, z_traj);

  // 7. x_traj = z_traj @ W_dec + b_dec  (M=33024, K=256, N=64)
  gemm_wmma<<<dim3(1, 258, 1), 256, 0, stream>>>(
      z_traj, 0, 256,  W_dec, 0, 64,  x_traj, 0, 64,  b_dec,
      33024, 64, 256, 1.f, 1.f, 0, nullptr, 0, 0.f, nullptr, 0);

  // 8. x_traj_gt = concat(x_init, x_future)
  gt_copy_kernel<<<(256 * 129 * 64 + 255) / 256, 256, 0, stream>>>(
      x_init, x_future, x_gt);

  // 9. z_traj_re = x_traj @ W_enc + b_enc  (M=33024, K=64, N=256)
  gemm_wmma<<<dim3(2, 258, 1), 256, 0, stream>>>(
      x_traj, 0, 64,  W_enc, 0, 256,  z_re, 0, 256,  b_enc,
      33024, 256, 64, 1.f, 1.f, 0, nullptr, 0, 0.f, nullptr, 0);

  // 10. z_traj_gt = x_traj_gt @ W_enc + b_enc
  gemm_wmma<<<dim3(2, 258, 1), 256, 0, stream>>>(
      x_gt, 0, 64,  W_enc, 0, 256,  z_gt, 0, 256,  b_enc,
      33024, 256, 64, 1.f, 1.f, 0, nullptr, 0, 0.f, nullptr, 0);
}
